// AtomDMPNN_20469814133012
// MI455X (gfx1250) — matrix operand order repaired and checked by reading, not compiled, not run on hardware
//
#include <hip/hip_runtime.h>
#include <hip/hip_bf16.h>

// ---------------- problem constants ----------------
#define NNODES 20000
#define NEDGES 640000
#define HID    128
#define EDIM   64

typedef __attribute__((ext_vector_type(16))) __bf16 bf16x16;
typedef __attribute__((ext_vector_type(8)))  float  floatx8;

union ABFrag {
    bf16x16        v;
    unsigned int   w[8];
    unsigned short u[16];
    uint4          q[2];
};

// pack two f32 into one dword of bf16 (lo = a, hi = b), round-half-up via +0x8000,
// then a single v_perm_b32 grabs the two high halves.
__device__ __forceinline__ unsigned int pk_bf16(float a, float b) {
    unsigned int ua = __builtin_bit_cast(unsigned int, a) + 0x8000u;
    unsigned int ub = __builtin_bit_cast(unsigned int, b) + 0x8000u;
    return __builtin_amdgcn_perm(ub, ua, 0x07060302u);  // {b.hi16, a.hi16}
}

__device__ __forceinline__ unsigned short f2bf(float f) {
    unsigned int u = __builtin_bit_cast(unsigned int, f);
    unsigned int r = u + 0x7FFFu + ((u >> 16) & 1u);    // RNE (pack kernel only)
    return (unsigned short)(r >> 16);
}

// K offset pattern for 16-bit A-matrix 16x32 fragment (ISA 7.12.2):
// lanes 0-15 (h=0): VGPR0..3 -> K=0..7 ; VGPR4..7 -> K=16..23
// lanes 16-31 (h=1): K=8..15 ; K=24..31
__device__ __forceinline__ int apatk(int i, int h) {
    return (i < 4) ? (2 * i + 8 * h) : (16 + 2 * (i - 4) + 8 * h);
}

__device__ __forceinline__ bf16x16 load_bfrag(const uint4* __restrict__ pb, int frag, int lane) {
    ABFrag t;
    const uint4* p = pb + ((size_t)frag * 32 + lane) * 2;
    t.q[0] = p[0];
    t.q[1] = p[1];
    return t.v;
}

__device__ __forceinline__ floatx8 wmma_bf16(bf16x16 a, bf16x16 b, floatx8 c) {
    return __builtin_amdgcn_wmma_f32_16x16x32_bf16(false, a, false, b, (short)0, c, false, false);
}

// ---------------- weight packing: W [Nn rows, Kk cols] row-major -> B fragments ----------------
// B matrix is W^T (Kk x Nn). Fragment (kt,nt): 32x16 bf16 tile, lane-major:
// lane l (h=l>>4, n=nt*16+(l&15)), uint j: bf16 pair at K = kt*32 + apatk(j, h).
__global__ void pack_b_kernel(const float* __restrict__ W, unsigned int* __restrict__ out,
                              int Kk, int Nn) {
    int ntCount = Nn >> 4;
    int total   = (Kk >> 5) * ntCount * 32 * 8;    // uints
    for (int t = blockIdx.x * blockDim.x + threadIdx.x; t < total;
         t += gridDim.x * blockDim.x) {
        int j    = t & 7;
        int lane = (t >> 3) & 31;
        int f    = t >> 8;
        int kt   = f / ntCount;
        int nt   = f - kt * ntCount;
        int h    = lane >> 4;
        int n    = nt * 16 + (lane & 15);
        int K    = kt * 32 + apatk(j, h);
        unsigned short lo = f2bf(W[(size_t)n * Kk + K]);
        unsigned short hi = f2bf(W[(size_t)n * Kk + K + 1]);
        out[t] = (unsigned int)lo | ((unsigned int)hi << 16);
    }
}

// ---------------- f32 -> bf16 row conversion (2 elems / thread via v_perm) ----------------
__global__ void cvt_bf16_kernel(const float* __restrict__ src, unsigned int* __restrict__ dst,
                                int npairs) {
    for (int i = blockIdx.x * blockDim.x + threadIdx.x; i < npairs; i += gridDim.x * blockDim.x) {
        float2 p = *(const float2*)(src + 2 * (size_t)i);
        dst[i] = pk_bf16(p.x, p.y);
    }
}

__global__ void zero_kernel(float* __restrict__ p, int n) {
    for (int i = blockIdx.x * blockDim.x + threadIdx.x; i < n; i += gridDim.x * blockDim.x)
        p[i] = 0.0f;
}

// ---------------- edge message GEMM + scatter-add ----------------
// One wave per 32-edge tile (two 16-row A subtiles sharing every B fragment).
// A = [x[src] | edge_attr] (bf16), B = W_msg^T (192x128). 96 WMMAs per wave;
// scatter via global_atomic_add_f32 into agg (L2 resident).
__global__ __launch_bounds__(128) void edge_msg_kernel(
    const unsigned short* __restrict__ xb,        // [N,128] bf16
    const float* __restrict__ edge_attr,          // [E,64] f32
    const long long* __restrict__ edge_index,     // [2,E] int64
    const uint4* __restrict__ pbW,                // packed W_msg^T frags (kt*8+nt)
    float* __restrict__ agg)                      // [N,128] f32
{
    const int lane = threadIdx.x & 31;
    const int wave = threadIdx.x >> 5;
    const int et2  = blockIdx.x * 4 + wave;       // 0..19999 (exact)
    const int m    = lane & 15;
    const int h    = lane >> 4;
    const size_t base = (size_t)et2 * 32;
    const size_t e0 = base + m;
    const size_t e1 = base + 16 + m;

    const int s0 = (int)edge_index[e0];
    const int s1 = (int)edge_index[e1];

    // ---- A fragments for both subtiles ----
    bf16x16 afr0[6], afr1[6];
    {
        const unsigned short* xr0 = xb + (size_t)s0 * HID;
        const unsigned short* xr1 = xb + (size_t)s1 * HID;
        #pragma unroll
        for (int kt = 0; kt < 4; ++kt) {
            ABFrag t0, t1;
            #pragma unroll
            for (int i = 0; i < 8; ++i) {
                int K = kt * 32 + apatk(i, h);
                t0.w[i] = *(const unsigned int*)(xr0 + K);   // bf16 pair
                t1.w[i] = *(const unsigned int*)(xr1 + K);
            }
            afr0[kt] = t0.v;
            afr1[kt] = t1.v;
        }
        const float* ea0 = edge_attr + e0 * EDIM;
        const float* ea1 = edge_attr + e1 * EDIM;
        #pragma unroll
        for (int kt = 0; kt < 2; ++kt) {
            ABFrag t0, t1;
            #pragma unroll
            for (int i = 0; i < 8; ++i) {
                int K = kt * 32 + apatk(i, h);
                float2 p0 = *(const float2*)(ea0 + K);
                float2 p1 = *(const float2*)(ea1 + K);
                t0.w[i] = pk_bf16(p0.x, p0.y);
                t1.w[i] = pk_bf16(p1.x, p1.y);
            }
            afr0[4 + kt] = t0.v;
            afr1[4 + kt] = t1.v;
        }
    }

    // destination nodes for the 8 C rows this lane owns (rows 8h..8h+7) per subtile
    int dst0[8], dst1[8];
    #pragma unroll
    for (int g = 0; g < 8; ++g) {
        dst0[g] = (int)edge_index[(size_t)NEDGES + base + 8 * h + g];
        dst1[g] = (int)edge_index[(size_t)NEDGES + base + 16 + 8 * h + g];
    }

    #pragma unroll
    for (int nt = 0; nt < 8; ++nt) {
        floatx8 acc0 = {}, acc1 = {};
        #pragma unroll
        for (int kt = 0; kt < 6; ++kt) {
            bf16x16 b = load_bfrag(pbW, kt * 8 + nt, lane);
            acc0 = wmma_bf16(afr0[kt], b, acc0);
            acc1 = wmma_bf16(afr1[kt], b, acc1);
        }
        const int col = nt * 16 + m;              // C: N = lane&15
        #pragma unroll
        for (int g = 0; g < 8; ++g) {             // C: M = g + 8h
            unsafeAtomicAdd(&agg[(size_t)dst0[g] * HID + col], acc0[g]);
            unsafeAtomicAdd(&agg[(size_t)dst1[g] * HID + col], acc1[g]);
        }
    }
}

// ---------------- GRU cell ----------------
// One wave per 16-node tile (1250 tiles). gi = agg@Wih^T + bih ; gh = h@Whh^T + bhh.
// Loop over 8 output column blocks; 6 live accumulators (r/z/n x ih/hh).
__global__ __launch_bounds__(64) void gru_kernel(
    const float* __restrict__ aggm,               // [N,128] message input
    const float* __restrict__ hprev,              // [N,128]
    const uint4* __restrict__ pbWih,              // frags (kt*24+nt), Nn=384
    const uint4* __restrict__ pbWhh,
    const float* __restrict__ bih,
    const float* __restrict__ bhh,
    float* __restrict__ hout)                     // [N,128]
{
    const int lane = threadIdx.x & 31;
    const int wave = threadIdx.x >> 5;
    const int t    = blockIdx.x * 2 + wave;       // 0..1249 (exact)
    const int m    = lane & 15;
    const int h    = lane >> 4;
    const size_t rowbase = (size_t)(t * 16 + m) * HID;

    // A fragments: message (agg) and hidden state (hprev), f32 -> bf16 via v_perm
    bf16x16 aM[4], aH[4];
    #pragma unroll
    for (int kt = 0; kt < 4; ++kt) {
        ABFrag tm, th;
        #pragma unroll
        for (int i = 0; i < 8; ++i) {
            int K = kt * 32 + apatk(i, h);
            float2 pm = *(const float2*)(aggm + rowbase + K);
            float2 ph = *(const float2*)(hprev + rowbase + K);
            tm.w[i] = pk_bf16(pm.x, pm.y);
            th.w[i] = pk_bf16(ph.x, ph.y);
        }
        aM[kt] = tm.v;
        aH[kt] = th.v;
    }

    #pragma unroll 1
    for (int cb = 0; cb < 8; ++cb) {              // output column block
        floatx8 aIr = {}, aIz = {}, aIn = {}, aHr = {}, aHz = {}, aHn = {};
        #pragma unroll
        for (int kt = 0; kt < 4; ++kt) {
            bf16x16 b;
            b = load_bfrag(pbWih, kt * 24 +      cb, lane); aIr = wmma_bf16(aM[kt], b, aIr);
            b = load_bfrag(pbWih, kt * 24 +  8 + cb, lane); aIz = wmma_bf16(aM[kt], b, aIz);
            b = load_bfrag(pbWih, kt * 24 + 16 + cb, lane); aIn = wmma_bf16(aM[kt], b, aIn);
            b = load_bfrag(pbWhh, kt * 24 +      cb, lane); aHr = wmma_bf16(aH[kt], b, aHr);
            b = load_bfrag(pbWhh, kt * 24 +  8 + cb, lane); aHz = wmma_bf16(aH[kt], b, aHz);
            b = load_bfrag(pbWhh, kt * 24 + 16 + cb, lane); aHn = wmma_bf16(aH[kt], b, aHn);
        }
        const int col = cb * 16 + m;
        const float bir = bih[col], biz = bih[col + HID], bin = bih[col + 2 * HID];
        const float bhr = bhh[col], bhz = bhh[col + HID], bhn = bhh[col + 2 * HID];
        #pragma unroll
        for (int g = 0; g < 8; ++g) {
            const int row = t * 16 + g + 8 * h;
            const float hv = hprev[(size_t)row * HID + col];
            const float r  = 1.0f / (1.0f + __expf(-((aIr[g] + bir) + (aHr[g] + bhr))));
            const float z  = 1.0f / (1.0f + __expf(-((aIz[g] + biz) + (aHz[g] + bhz))));
            const float n  = tanhf((aIn[g] + bin) + r * (aHn[g] + bhn));
            hout[(size_t)row * HID + col] = (1.0f - z) * n + z * hv;
        }
    }
}

// ---------------- host side ----------------
static inline size_t align256(size_t x) { return (x + 255) & ~(size_t)255; }

extern "C" void kernel_launch(void* const* d_in, const int* in_sizes, int n_in,
                              void* d_out, int out_size, void* d_ws, size_t ws_size,
                              hipStream_t stream) {
    const float*     x       = (const float*)d_in[0];
    const long long* eidx    = (const long long*)d_in[1];
    const float*     eattr   = (const float*)d_in[2];
    const float*     Wmsg1   = (const float*)d_in[3];
    const float*     Wih1    = (const float*)d_in[4];
    const float*     Whh1    = (const float*)d_in[5];
    const float*     bih1    = (const float*)d_in[6];
    const float*     bhh1    = (const float*)d_in[7];
    const float*     Wmsg2   = (const float*)d_in[8];
    const float*     Wih2    = (const float*)d_in[9];
    const float*     Whh2    = (const float*)d_in[10];
    const float*     bih2    = (const float*)d_in[11];
    const float*     bhh2    = (const float*)d_in[12];
    float*           out     = (float*)d_out;

    // workspace carve-up
    char* ws = (char*)d_ws;
    size_t off = 0;
    unsigned int* pbW1   = (unsigned int*)(ws + off); off = align256(off + 192 * 128 * 2);
    unsigned int* pbW2   = (unsigned int*)(ws + off); off = align256(off + 192 * 128 * 2);
    unsigned int* pbWih1 = (unsigned int*)(ws + off); off = align256(off + 128 * 384 * 2);
    unsigned int* pbWhh1 = (unsigned int*)(ws + off); off = align256(off + 128 * 384 * 2);
    unsigned int* pbWih2 = (unsigned int*)(ws + off); off = align256(off + 128 * 384 * 2);
    unsigned int* pbWhh2 = (unsigned int*)(ws + off); off = align256(off + 128 * 384 * 2);
    unsigned short* xb   = (unsigned short*)(ws + off); off = align256(off + (size_t)NNODES * HID * 2);
    float* agg           = (float*)(ws + off);          off = align256(off + (size_t)NNODES * HID * 4);
    float* x1            = (float*)(ws + off);          off = align256(off + (size_t)NNODES * HID * 4);
    (void)off; (void)ws_size; (void)n_in; (void)in_sizes; (void)out_size;

    const int nfeat  = NNODES * HID;
    const int npairs = nfeat / 2;

    // pack all weights to bf16 B-fragment layout (L2-hot, reused by every wave)
    pack_b_kernel<<<48,  256, 0, stream>>>(Wmsg1, pbW1,   192, 128);
    pack_b_kernel<<<48,  256, 0, stream>>>(Wmsg2, pbW2,   192, 128);
    pack_b_kernel<<<96,  256, 0, stream>>>(Wih1,  pbWih1, 128, 384);
    pack_b_kernel<<<96,  256, 0, stream>>>(Whh1,  pbWhh1, 128, 384);
    pack_b_kernel<<<96,  256, 0, stream>>>(Wih2,  pbWih2, 128, 384);
    pack_b_kernel<<<96,  256, 0, stream>>>(Whh2,  pbWhh2, 128, 384);

    // ---- layer 1 ----
    cvt_bf16_kernel<<<(npairs + 255) / 256, 256, 0, stream>>>(x, (unsigned int*)xb, npairs);
    zero_kernel<<<(nfeat + 255) / 256, 256, 0, stream>>>(agg, nfeat);
    edge_msg_kernel<<<NEDGES / 128, 128, 0, stream>>>((const unsigned short*)xb, eattr, eidx,
                                                      (const uint4*)pbW1, agg);
    gru_kernel<<<NNODES / 32, 64, 0, stream>>>(agg, x, (const uint4*)pbWih1,
                                               (const uint4*)pbWhh1, bih1, bhh1, x1);

    // ---- layer 2 ----
    cvt_bf16_kernel<<<(npairs + 255) / 256, 256, 0, stream>>>(x1, (unsigned int*)xb, npairs);
    zero_kernel<<<(nfeat + 255) / 256, 256, 0, stream>>>(agg, nfeat);
    edge_msg_kernel<<<NEDGES / 128, 128, 0, stream>>>((const unsigned short*)xb, eattr, eidx,
                                                      (const uint4*)pbW2, agg);
    gru_kernel<<<NNODES / 32, 64, 0, stream>>>(agg, x1, (const uint4*)pbWih2,
                                               (const uint4*)pbWhh2, bih2, bhh2, out);
}
